// MultiHead_Attention_16020228014689
// MI455X (gfx1250) — matrix operand order repaired and checked by reading, not compiled, not run on hardware
//
#include <hip/hip_runtime.h>
#include <hip/hip_bf16.h>

typedef __attribute__((ext_vector_type(16))) __bf16 v16bf;
typedef __attribute__((ext_vector_type(8)))  float  v8f;

#define D_MODEL   512
#define QKV_DIM   1536
#define TILE_M    32          // tokens per workgroup (2 M-tiles of 16)
#define NTHREADS  512         // 16 waves
#define KSTEPS    16          // 512 / 32
#define QKV_NT    96          // 1536 / 16
#define OUT_NT    32          // 512 / 16
#define QKV_STRIDE 1540       // padded f32 row stride (8*1540 % 64 == 32 -> conflict-free C stores)

#define WQ_FRAG_ELEMS (QKV_NT * KSTEPS * 32 * 16)   // 786432 bf16
#define WO_FRAG_ELEMS (OUT_NT * KSTEPS * 32 * 16)   // 262144 bf16

// ---------------------------------------------------------------------------
// Fragment layouts (CDNA5 ISA 7.12.2, wave32):
// 16-bit A 16x32:  lane = m + 16*(kk>>4),      vec idx = kk & 15   (contiguous in K; not ISA table order)
// 16-bit B 32x16:  lane = n + 16*(k>>4),      vec idx = k & 15   (contiguous in K; not ISA table order)
// 32-bit C 16x16:  vgpr r: lanes 0-15 -> (M=r, N=lane), lanes 16-31 -> (M=r+8)
// LDS A-frag array: [ks][mtile][lane][16]  (two 16-row M-tiles per block)
// ---------------------------------------------------------------------------
__device__ __forceinline__ int a_frag_off(int m, int k) {
    int ks  = k >> 5;
    int kk  = k & 31;
    int sel = kk >> 4;
    int mt  = m >> 4;
    int ln  = (m & 15) + 16 * sel;
    int idx = kk & 15;
    return ((ks * 2 + mt) * 32 + ln) * 16 + idx;
}

__device__ __forceinline__ void split_bf16(float v, __bf16& h, __bf16& l) {
    h = (__bf16)v;
    l = (__bf16)(v - (float)h);
}

#define WMMA(A, B, C) __builtin_amdgcn_wmma_f32_16x16x32_bf16(false, (A), false, (B), (short)0, (C), false, false)

// ---------------------------------------------------------------------------
// Prep: f32 weights (row-major [e][d]) -> hi/lo bf16 B-fragments in ws.
// One thread = one B-fragment lane: 16 contiguous f32 in, 16+16 bf16 out.
// ---------------------------------------------------------------------------
__global__ void mha_prep_weights(const float* __restrict__ wq, const float* __restrict__ wo,
                                 __bf16* __restrict__ qh, __bf16* __restrict__ ql,
                                 __bf16* __restrict__ oh, __bf16* __restrict__ ol) {
    int t = blockIdx.x * blockDim.x + threadIdx.x;
    const float* src;
    __bf16 *dh, *dl;
    if (t < QKV_NT * KSTEPS * 32) {
        int lane = t & 31, frag = t >> 5;
        int ks = frag & (KSTEPS - 1), nt = frag >> 4;
        src = wq + (nt * 16 + (lane & 15)) * D_MODEL + ks * 32 + (lane >> 4) * 16;
        dh = qh + t * 16; dl = ql + t * 16;
    } else {
        int u = t - QKV_NT * KSTEPS * 32;
        if (u >= OUT_NT * KSTEPS * 32) return;
        int lane = u & 31, frag = u >> 5;
        int ks = frag & (KSTEPS - 1), nt = frag >> 4;
        src = wo + (nt * 16 + (lane & 15)) * D_MODEL + ks * 32 + (lane >> 4) * 16;
        dh = oh + u * 16; dl = ol + u * 16;
    }
#pragma unroll
    for (int j = 0; j < 16; ++j) {
        __bf16 h, l;
        split_bf16(src[j], h, l);
        dh[j] = h; dl[j] = l;
    }
}

// ---------------------------------------------------------------------------
// Fused kernel: 1 block = 32 tokens, 512 threads = 16 waves.
// Each wave computes 2M x 2N register-blocked tile groups (12 WMMA / k-step).
// ---------------------------------------------------------------------------
__global__ void __launch_bounds__(NTHREADS)
mha_fused(const float* __restrict__ x,
          const float* __restrict__ bqkv,
          const float* __restrict__ bout,
          const __bf16* __restrict__ wqh, const __bf16* __restrict__ wql,
          const __bf16* __restrict__ woh, const __bf16* __restrict__ wol,
          float* __restrict__ out) {
    extern __shared__ __align__(32) char smem[];
    __bf16* ldsA_hi = (__bf16*)smem;                                   // 32 KB
    __bf16* ldsA_lo = ldsA_hi + TILE_M * D_MODEL;                      // 32 KB
    float*  qkvs    = (float*)(smem + 2 * TILE_M * D_MODEL * 2);       // 32*1540 f32 = 192.5 KB

    const int  tid  = threadIdx.x;
    const int  lane = tid & 31;
    const int  wv   = tid >> 5;                 // 0..15
    const long t0   = (long)blockIdx.x * TILE_M;

    // ---- Phase 1: x tile -> hi/lo bf16 A-fragments in LDS -----------------
    for (int i = tid; i < TILE_M * D_MODEL; i += NTHREADS) {
        int m = i >> 9;
        int k = i & (D_MODEL - 1);
        __bf16 h, l;
        split_bf16(x[(t0 + m) * D_MODEL + k], h, l);
        int off = a_frag_off(m, k);
        ldsA_hi[off] = h;
        ldsA_lo[off] = l;
    }
    __syncthreads();

    const v16bf* pAh = (const v16bf*)ldsA_hi;
    const v16bf* pAl = (const v16bf*)ldsA_lo;

    // ---- Phase 2: QKV GEMM (32x1536): 48 2N-units, 3 per wave -------------
    for (int u = 0; u < 3; ++u) {
        int nt0 = (wv * 3 + u) * 2;
        const v16bf* pB0h = (const v16bf*)wqh + (size_t)nt0 * KSTEPS * 32 + lane;
        const v16bf* pB0l = (const v16bf*)wql + (size_t)nt0 * KSTEPS * 32 + lane;
        const v16bf* pB1h = pB0h + KSTEPS * 32;
        const v16bf* pB1l = pB0l + KSTEPS * 32;
        __builtin_prefetch(pB0h + 2 * KSTEPS * 32, 0, 3);   // next unit's B
        v8f c00 = {}, c01 = {}, c10 = {}, c11 = {};
#pragma unroll 2
        for (int ks = 0; ks < KSTEPS; ++ks) {
            v16bf a0h = pAh[(ks * 2 + 0) * 32 + lane];
            v16bf a0l = pAl[(ks * 2 + 0) * 32 + lane];
            v16bf a1h = pAh[(ks * 2 + 1) * 32 + lane];
            v16bf a1l = pAl[(ks * 2 + 1) * 32 + lane];
            v16bf b0h = pB0h[ks * 32];
            v16bf b0l = pB0l[ks * 32];
            v16bf b1h = pB1h[ks * 32];
            v16bf b1l = pB1l[ks * 32];
            c00 = WMMA(a0h, b0h, c00); c00 = WMMA(a0h, b0l, c00); c00 = WMMA(a0l, b0h, c00);
            c01 = WMMA(a0h, b1h, c01); c01 = WMMA(a0h, b1l, c01); c01 = WMMA(a0l, b1h, c01);
            c10 = WMMA(a1h, b0h, c10); c10 = WMMA(a1h, b0l, c10); c10 = WMMA(a1l, b0h, c10);
            c11 = WMMA(a1h, b1h, c11); c11 = WMMA(a1h, b1l, c11); c11 = WMMA(a1l, b1h, c11);
        }
        int n = lane & 15, g2 = lane >> 4;
        float bias0 = bqkv[nt0 * 16 + n];
        float bias1 = bqkv[(nt0 + 1) * 16 + n];
#pragma unroll
        for (int r = 0; r < 8; ++r) {
            int mr = r + 8 * g2;
            qkvs[mr * QKV_STRIDE + nt0 * 16 + n]              = c00[r] + bias0;
            qkvs[mr * QKV_STRIDE + (nt0 + 1) * 16 + n]        = c01[r] + bias1;
            qkvs[(16 + mr) * QKV_STRIDE + nt0 * 16 + n]       = c10[r] + bias0;
            qkvs[(16 + mr) * QKV_STRIDE + (nt0 + 1) * 16 + n] = c11[r] + bias1;
        }
    }
    __syncthreads();

    // ---- Phase 3: per-token attention over heads (32 tokens x 8 heads) ----
    if (tid < TILE_M * 8) {
        int m = tid >> 3;
        int h = tid & 7;
        const float* row = qkvs + m * QKV_STRIDE;   // qkv[t,e], e = g*192 + {q:0,k:64,v:128} + d
        float q[64];
#pragma unroll
        for (int d = 0; d < 64; ++d) q[d] = row[h * 192 + d];

        float s[8];
        float mx = -3.0e38f;
#pragma unroll
        for (int g = 0; g < 8; ++g) {
            float a = 0.f;
#pragma unroll
            for (int d = 0; d < 64; ++d) a += q[d] * row[g * 192 + 64 + d];
            a *= 0.125f;                            // 1/sqrt(64)
            s[g] = a;
            mx = fmaxf(mx, a);
        }
        float sum = 0.f;
#pragma unroll
        for (int g = 0; g < 8; ++g) { s[g] = __expf(s[g] - mx); sum += s[g]; }
        float inv = 1.0f / sum;
#pragma unroll
        for (int g = 0; g < 8; ++g) s[g] *= inv;

        // val[m, h*64+d] -> hi/lo bf16 A-fragments (reuse ldsA after barrier)
#pragma unroll
        for (int d = 0; d < 64; ++d) {
            float a = 0.f;
#pragma unroll
            for (int g = 0; g < 8; ++g) a += s[g] * row[g * 192 + 128 + d];
            __bf16 hh, ll;
            split_bf16(a, hh, ll);
            int off = a_frag_off(m, h * 64 + d);
            ldsA_hi[off] = hh;
            ldsA_lo[off] = ll;
        }
    }
    __syncthreads();

    // ---- Phase 4: output GEMM (32x512): 16 2N-units, 1 per wave -----------
    {
        int nt0 = wv * 2;
        const v16bf* pB0h = (const v16bf*)woh + (size_t)nt0 * KSTEPS * 32 + lane;
        const v16bf* pB0l = (const v16bf*)wol + (size_t)nt0 * KSTEPS * 32 + lane;
        const v16bf* pB1h = pB0h + KSTEPS * 32;
        const v16bf* pB1l = pB0l + KSTEPS * 32;
        v8f c00 = {}, c01 = {}, c10 = {}, c11 = {};
#pragma unroll 2
        for (int ks = 0; ks < KSTEPS; ++ks) {
            v16bf a0h = pAh[(ks * 2 + 0) * 32 + lane];
            v16bf a0l = pAl[(ks * 2 + 0) * 32 + lane];
            v16bf a1h = pAh[(ks * 2 + 1) * 32 + lane];
            v16bf a1l = pAl[(ks * 2 + 1) * 32 + lane];
            v16bf b0h = pB0h[ks * 32];
            v16bf b0l = pB0l[ks * 32];
            v16bf b1h = pB1h[ks * 32];
            v16bf b1l = pB1l[ks * 32];
            c00 = WMMA(a0h, b0h, c00); c00 = WMMA(a0h, b0l, c00); c00 = WMMA(a0l, b0h, c00);
            c01 = WMMA(a0h, b1h, c01); c01 = WMMA(a0h, b1l, c01); c01 = WMMA(a0l, b1h, c01);
            c10 = WMMA(a1h, b0h, c10); c10 = WMMA(a1h, b0l, c10); c10 = WMMA(a1l, b0h, c10);
            c11 = WMMA(a1h, b1h, c11); c11 = WMMA(a1h, b1l, c11); c11 = WMMA(a1l, b1h, c11);
        }
        int n = lane & 15, g2 = lane >> 4;
        float bias0 = bout[nt0 * 16 + n];
        float bias1 = bout[(nt0 + 1) * 16 + n];
#pragma unroll
        for (int r = 0; r < 8; ++r) {
            int mr = r + 8 * g2;
            out[(t0 + mr) * D_MODEL + nt0 * 16 + n]              = c00[r] + bias0;
            out[(t0 + mr) * D_MODEL + (nt0 + 1) * 16 + n]        = c01[r] + bias1;
            out[(t0 + 16 + mr) * D_MODEL + nt0 * 16 + n]         = c10[r] + bias0;
            out[(t0 + 16 + mr) * D_MODEL + (nt0 + 1) * 16 + n]   = c11[r] + bias1;
        }
    }
}

// ---------------------------------------------------------------------------
extern "C" void kernel_launch(void* const* d_in, const int* in_sizes, int n_in,
                              void* d_out, int out_size, void* d_ws, size_t ws_size,
                              hipStream_t stream) {
    const float* x    = (const float*)d_in[0];
    const float* wqkv = (const float*)d_in[1];
    const float* bqkv = (const float*)d_in[2];
    const float* wout = (const float*)d_in[3];
    const float* bout = (const float*)d_in[4];
    float*       out  = (float*)d_out;

    __bf16* qh = (__bf16*)d_ws;
    __bf16* ql = (__bf16*)((char*)d_ws + (size_t)WQ_FRAG_ELEMS * 2);
    __bf16* oh = (__bf16*)((char*)d_ws + (size_t)WQ_FRAG_ELEMS * 4);
    __bf16* ol = (__bf16*)((char*)d_ws + (size_t)WQ_FRAG_ELEMS * 4 + (size_t)WO_FRAG_ELEMS * 2);

    int n_tokens = in_sizes[0] / D_MODEL;                 // B*L = 61440
    int prep_threads = (QKV_NT + OUT_NT) * KSTEPS * 32;   // 65536
    mha_prep_weights<<<prep_threads / 256, 256, 0, stream>>>(wqkv, wout, qh, ql, oh, ol);

    size_t shmem = (size_t)2 * TILE_M * D_MODEL * 2 + (size_t)TILE_M * QKV_STRIDE * 4; // 262656 B
    hipFuncSetAttribute((const void*)mha_fused,
                        hipFuncAttributeMaxDynamicSharedMemorySize, (int)shmem);
    mha_fused<<<n_tokens / TILE_M, NTHREADS, shmem, stream>>>(x, bqkv, bout, qh, ql, oh, ol, out);
}
